// ClassAttention_51840255262861
// MI455X (gfx1250) — matrix operand order repaired and checked by reading, not compiled, not run on hardware
//
#include <hip/hip_runtime.h>
#include <math.h>

// ClassAttention fused kernel for gfx1250 (MI455X), wave32, WMMA bf16x3 split.
//
// Pipeline:
//   K1 qproj   : qs[b,h,d]   = scale * x[b,0,:] @ Wq[:, h*64+d]          (fp32 VALU, tiny)
//   K2 kvattn  : per (b, 128-row tile, head): K/V tile = x_tile @ Wkv via
//                v_wmma_f32_16x16x32_bf16 (hi/lo split, 3 wmma per tile),
//                then online-softmax partial {m, l, sum(p*V)[64]} -> ws
//   K3 combine : log-sum-exp combine of 32 tile-partials -> cls_embed[b,1024]
//   K4 proj    : out = cls_embed @ Wproj + bproj                          (fp32 VALU)

typedef __attribute__((ext_vector_type(16))) __bf16 v16bf;
typedef __attribute__((ext_vector_type(8)))  __bf16 v8bf;
typedef __attribute__((ext_vector_type(8)))  float  v8f;

union V16U { v16bf v; v8bf h[2]; };

__device__ __forceinline__ void bf16_split(float f, __bf16& hi, __bf16& lo) {
    hi = (__bf16)f;
    lo = (__bf16)(f - (float)hi);
}

// ---------------------------------------------------------------- K1: q projection
// grid (16 b, 16 h), 256 threads. qs[(b*16+h)*64 + d]
__global__ void __launch_bounds__(256) ca_qproj(const float* __restrict__ x,
                                                const float* __restrict__ Wq,
                                                float* __restrict__ qs) {
    const int b = blockIdx.x, h = blockIdx.y;
    const int tid = threadIdx.x;
    const int d = tid & 63, part = tid >> 6;         // 4 K-parts of 256
    const float* xr = x + (size_t)b * 4096 * 1024;   // row 0 of batch b
    float s = 0.f;
    const int c0 = part * 256;
    for (int c = c0; c < c0 + 256; ++c)
        s += xr[c] * Wq[(size_t)c * 1024 + h * 64 + d];
    __shared__ float red[4][64];
    red[part][d] = s;
    __syncthreads();
    if (tid < 64) {
        float v = red[0][tid] + red[1][tid] + red[2][tid] + red[3][tid];
        qs[(b * 16 + h) * 64 + tid] = v * 0.125f;    // 1/sqrt(64)
    }
}

// ---------------------------------------------------------------- K2: fused kv GEMM + attention partials
// grid (16 b, 32 tiles), 256 threads (8 waves). TM=128 rows, BK=32 K-chunk.
// partial record (b,h,t): 66 floats {m, l, acc[64]} at part[((b*16+h)*32+t)*66]

#define TM 128
#define BK 32
#define SX_STRIDE 69   // 128x64 K-tile, stride 69 -> bank = (5n+d)%64, conflict-free

struct GemmBufs {
    __bf16 Axh[TM][BK];
    __bf16 Axl[TM][BK];
    __bf16 Bwh[128][BK];   // Wkv chunk, transposed: [col][k], cols 0..63=K-head, 64..127=V-head
    __bf16 Bwl[128][BK];
};
union SMemB {
    GemmBufs g;                 // 32 KB, live only during GEMM
    float    Sx[TM * SX_STRIDE]; // 35.3 KB, live only during attention phase
};

__global__ void __launch_bounds__(256) ca_kvattn(const float* __restrict__ x,
                                                 const float* __restrict__ Wkv,
                                                 const float* __restrict__ qs,
                                                 float* __restrict__ part) {
    const int b   = blockIdx.x;
    const int t   = blockIdx.y;
    const int tid = threadIdx.x;
    const int wave = tid >> 5;
    const int lane = tid & 31;
    const int g    = lane >> 4;      // half-wave
    const int ln   = lane & 15;

    __shared__ __align__(16) SMemB sm;
    __shared__ float q_s[64];
    __shared__ float red[128];
    __shared__ float vred[16][64];
    __shared__ float MLs[2];

    const float* xb = x + ((size_t)b * 4096 + (size_t)t * TM) * 1024;

    for (int h = 0; h < 16; ++h) {
        if (tid < 64) q_s[tid] = qs[(b * 16 + h) * 64 + tid];

        v8f acc[8];
        #pragma unroll
        for (int c = 0; c < 8; ++c)
            #pragma unroll
            for (int j = 0; j < 8; ++j) acc[c][j] = 0.f;

        for (int k0 = 0; k0 < 1024; k0 += BK) {
            __syncthreads();   // prior reads of LDS buffers done
            // ---- stage x chunk (128 x 32 f32 -> bf16 hi/lo), 4 float4 per thread
            #pragma unroll
            for (int i = 0; i < 4; ++i) {
                int idx = tid + (i << 8);
                int r  = idx >> 3;
                int c4 = (idx & 7) << 2;
                const float4 v = *(const float4*)(xb + (size_t)r * 1024 + k0 + c4);
                __bf16 hh, ll;
                bf16_split(v.x, hh, ll); sm.g.Axh[r][c4 + 0] = hh; sm.g.Axl[r][c4 + 0] = ll;
                bf16_split(v.y, hh, ll); sm.g.Axh[r][c4 + 1] = hh; sm.g.Axl[r][c4 + 1] = ll;
                bf16_split(v.z, hh, ll); sm.g.Axh[r][c4 + 2] = hh; sm.g.Axl[r][c4 + 2] = ll;
                bf16_split(v.w, hh, ll); sm.g.Axh[r][c4 + 3] = hh; sm.g.Axl[r][c4 + 3] = ll;
            }
            // ---- stage Wkv chunk (32 k x 128 cols), transposed to [col][k]
            // cols 0..63 -> Wkv[:, h*64..], cols 64..127 -> Wkv[:, 1024+h*64..]
            #pragma unroll
            for (int i = 0; i < 4; ++i) {
                int idx  = tid + (i << 8);
                int k    = idx >> 5;          // 0..31
                int j    = idx & 31;
                int half = j >> 4;
                int cc   = (j & 15) << 2;
                const float4 v = *(const float4*)(Wkv + (size_t)(k0 + k) * 2048 + half * 1024 + h * 64 + cc);
                int lc = half * 64 + cc;
                __bf16 hh, ll;
                bf16_split(v.x, hh, ll); sm.g.Bwh[lc + 0][k] = hh; sm.g.Bwl[lc + 0][k] = ll;
                bf16_split(v.y, hh, ll); sm.g.Bwh[lc + 1][k] = hh; sm.g.Bwl[lc + 1][k] = ll;
                bf16_split(v.z, hh, ll); sm.g.Bwh[lc + 2][k] = hh; sm.g.Bwl[lc + 2][k] = ll;
                bf16_split(v.w, hh, ll); sm.g.Bwh[lc + 3][k] = hh; sm.g.Bwl[lc + 3][k] = ll;
            }
            __syncthreads();

            // ---- fragments + WMMA. A 16x32: lane K-set = g*8+{0..7} U g*8+16+{0..7}
            const int mrow = (wave << 4) + ln;
            V16U Ah, Al;
            Ah.h[0] = *(const v8bf*)&sm.g.Axh[mrow][g * 8];
            Ah.h[1] = *(const v8bf*)&sm.g.Axh[mrow][g * 8 + 16];
            Al.h[0] = *(const v8bf*)&sm.g.Axl[mrow][g * 8];
            Al.h[1] = *(const v8bf*)&sm.g.Axl[mrow][g * 8 + 16];
            #pragma unroll
            for (int c = 0; c < 8; ++c) {
                const int col = (c << 4) + ln;      // B 32x16: lane K-set = g*8+{0..7} U g*8+16+{0..7}
                V16U Bh, Bl;
                Bh.h[0] = *(const v8bf*)&sm.g.Bwh[col][g * 8];
                Bh.h[1] = *(const v8bf*)&sm.g.Bwh[col][g * 8 + 16];
                Bl.h[0] = *(const v8bf*)&sm.g.Bwl[col][g * 8];
                Bl.h[1] = *(const v8bf*)&sm.g.Bwl[col][g * 8 + 16];
                acc[c] = __builtin_amdgcn_wmma_f32_16x16x32_bf16(false, Ah.v, false, Bh.v,
                                                                 (short)0, acc[c], false, false);
                acc[c] = __builtin_amdgcn_wmma_f32_16x16x32_bf16(false, Ah.v, false, Bl.v,
                                                                 (short)0, acc[c], false, false);
                acc[c] = __builtin_amdgcn_wmma_f32_16x16x32_bf16(false, Al.v, false, Bh.v,
                                                                 (short)0, acc[c], false, false);
            }
        }

        __syncthreads();   // all fragment reads done before Sx aliases the GEMM buffers

        // ---- spill K columns (c=0..3) to LDS; V columns (c=4..7) stay in registers
        #pragma unroll
        for (int c = 0; c < 4; ++c) {
            const int col = (c << 4) + ln;
            #pragma unroll
            for (int j = 0; j < 8; ++j) {
                const int row = (wave << 4) + j + (g << 3);
                sm.Sx[row * SX_STRIDE + col] = acc[c][j];
            }
        }
        __syncthreads();

        // ---- s_n = q . K_n  (128 rows, one thread each)
        if (tid < 128) {
            float s = 0.f;
            const float* kr = &sm.Sx[tid * SX_STRIDE];
            #pragma unroll 16
            for (int d = 0; d < 64; ++d) s += q_s[d] * kr[d];
            red[tid] = s;
        }
        __syncthreads();
        if (tid == 0) {
            float M = red[0];
            for (int n = 1; n < 128; ++n) M = fmaxf(M, red[n]);
            MLs[0] = M;
        }
        __syncthreads();
        const float M = MLs[0];
        if (tid < 128) red[tid] = __expf(red[tid] - M);
        __syncthreads();
        if (tid == 0) {
            float L = 0.f;
            for (int n = 0; n < 128; ++n) L += red[n];
            MLs[1] = L;
        }
        // ---- sum(p_n * V[n,d]) from accumulator registers (V rows owned per wave)
        #pragma unroll
        for (int cp = 0; cp < 4; ++cp) {
            float a = 0.f;
            #pragma unroll
            for (int j = 0; j < 8; ++j) {
                const int row = (wave << 4) + j + (g << 3);
                a += red[row] * acc[4 + cp][j];
            }
            vred[(wave << 1) + g][(cp << 4) + ln] = a;
        }
        __syncthreads();
        float* rec = part + (((size_t)b * 16 + h) * 32 + t) * 66;
        if (tid < 64) {
            float a = 0.f;
            #pragma unroll
            for (int s = 0; s < 16; ++s) a += vred[s][tid];
            rec[2 + tid] = a;
        }
        if (tid == 0) { rec[0] = M; rec[1] = MLs[1]; }
        __syncthreads();   // protect q_s / red / Sx reuse in next head
    }
}

// ---------------------------------------------------------------- K3: combine tile partials -> cls_embed
// grid (16 b), 256 threads
__global__ void __launch_bounds__(256) ca_combine(const float* __restrict__ part,
                                                  float* __restrict__ cls) {
    const int b = blockIdx.x, tid = threadIdx.x;
    __shared__ float wgt[16][32];
    __shared__ float Linv[16];
    if (tid < 16) {
        const int h = tid;
        const float* base = part + ((size_t)b * 16 + h) * 32 * 66;
        float M = -3.0e38f;
        for (int t = 0; t < 32; ++t) M = fmaxf(M, base[t * 66]);
        float L = 0.f;
        for (int t = 0; t < 32; ++t) {
            const float e = __expf(base[t * 66] - M);
            wgt[h][t] = e;
            L += e * base[t * 66 + 1];
        }
        Linv[h] = 1.0f / L;
    }
    __syncthreads();
    for (int i = 0; i < 4; ++i) {
        const int idx = tid + (i << 8);
        const int h = idx >> 6, d = idx & 63;
        const float* base = part + ((size_t)b * 16 + h) * 32 * 66 + 2 + d;
        float s = 0.f;
        for (int t = 0; t < 32; ++t) s += wgt[h][t] * base[t * 66];
        cls[b * 1024 + (h << 6) + d] = s * Linv[h];
    }
}

// ---------------------------------------------------------------- K4: output projection
// grid (16 b, 8 col-parts), 256 threads
__global__ void __launch_bounds__(256) ca_proj(const float* __restrict__ cls,
                                               const float* __restrict__ Wp,
                                               const float* __restrict__ bp,
                                               float* __restrict__ out) {
    const int b = blockIdx.x, cp = blockIdx.y, tid = threadIdx.x;
    __shared__ float c_s[1024];
    __shared__ float red[256];
    for (int i = tid; i < 1024; i += 256) c_s[i] = cls[b * 1024 + i];
    __syncthreads();
    const int col  = cp * 128 + (tid & 127);
    const int part = tid >> 7;              // 2 K-parts of 512
    float s = 0.f;
    const int i0 = part * 512;
    for (int i = i0; i < i0 + 512; ++i) s += c_s[i] * Wp[(size_t)i * 1024 + col];
    red[tid] = s;
    __syncthreads();
    if (tid < 128) out[b * 1024 + col] = red[tid] + red[tid + 128] + bp[col];
}

// ---------------------------------------------------------------- launcher
extern "C" void kernel_launch(void* const* d_in, const int* in_sizes, int n_in,
                              void* d_out, int out_size, void* d_ws, size_t ws_size,
                              hipStream_t stream) {
    (void)in_sizes; (void)n_in; (void)out_size; (void)ws_size;
    const float* x     = (const float*)d_in[0];
    const float* Wq    = (const float*)d_in[1];
    const float* Wkv   = (const float*)d_in[2];
    const float* Wproj = (const float*)d_in[3];
    const float* bproj = (const float*)d_in[4];
    float* out = (float*)d_out;

    // workspace layout (floats): qs[16*1024] | partials[16*16*32*66] | cls[16*1024]
    float* ws   = (float*)d_ws;
    float* qs   = ws;
    float* part = ws + 16384;
    float* cls  = ws + 16384 + 16 * 16 * 32 * 66;

    ca_qproj  <<<dim3(16, 16), 256, 0, stream>>>(x, Wq, qs);
    ca_kvattn <<<dim3(16, 32), 256, 0, stream>>>(x, Wkv, qs, part);
    ca_combine<<<dim3(16),     256, 0, stream>>>(part, cls);
    ca_proj   <<<dim3(16, 8),  256, 0, stream>>>(cls, Wproj, bproj, out);
}
